// BracketGNN_52578989637880
// MI455X (gfx1250) — matrix operand order repaired and checked by reading, not compiled, not run on hardware
//
#include <hip/hip_runtime.h>
#include <hip/hip_bf16.h>

// ---------------------------------------------------------------------------
// BracketGNN inference pipeline for MI455X (gfx1250, wave32, WMMA)
//   Per layer: GEMM[N,128]x[128,128] via v_wmma_f32_16x16x32_bf16 with
//   pre-swizzled bf16 fragments (pure global_load_b128 operand path) ->
//   edge scatter via global_atomic_add_f32 (+ global_prefetch_b8) ->
//   fused self-loop + bias + BN + ReLU epilogue that also emits the next
//   layer's bf16 WMMA-fragment operand.  Then mean pool -> classifier.
// ---------------------------------------------------------------------------

#define HDIM 128
#define CDIM 16
#define SHALF 64  // H/2

typedef __attribute__((ext_vector_type(16))) __bf16 v16bf;
typedef __attribute__((ext_vector_type(8)))  float  v8f;

// ------------------------------- utility -----------------------------------
__global__ void zero_f32(float* __restrict__ p, int n) {
    int i = blockIdx.x * blockDim.x + threadIdx.x;
    if (i < n) p[i] = 0.0f;
}

__global__ void set_ones_f32(float* __restrict__ p, int n) {
    int i = blockIdx.x * blockDim.x + threadIdx.x;
    if (i < n) p[i] = 1.0f;
}

// ------------------------- degree / normalization --------------------------
__global__ void deg_accum(const int* __restrict__ dst, float* __restrict__ deg, int E) {
    int e = blockIdx.x * blockDim.x + threadIdx.x;
    if (e < E) atomicAdd(&deg[dst[e]], 1.0f);
}

__global__ void finalize_dinv(float* __restrict__ deg, int n) {
    int i = blockIdx.x * blockDim.x + threadIdx.x;
    if (i < n) deg[i] = rsqrtf(fmaxf(deg[i], 1.0f));   // deg >= 1 via self-loop
}

// --------------------- WMMA fragment swizzle helpers ------------------------
// A-matrix 16x32 bf16 (ISA 7.12.2): lane = m | hi<<4,
//   ka = kk + (v>>2)*16 + hi*8 + (v&3)*2 + h,  e = 2v + h
// Inverse: given k32 = f & 31:
__device__ __forceinline__ size_t aswz_pos(int i, int f) {
    const int s   = f >> 5;            // K-step (0..3)
    const int k32 = f & 31;
    const int hi  = (k32 >> 3) & 1;
    const int v   = (((k32 >> 4) & 1) << 2) | ((k32 >> 1) & 3);
    const int e   = (v << 1) | (k32 & 1);
    const int lane = (i & 15) | (hi << 4);
    return ((((size_t)(i >> 4)) * 4 + s) * 32 + lane) * 16 + e;
}

// x -> Aswz (layer-0 operand prep)
__global__ void swizzle_x_kernel(const float* __restrict__ Hm,
                                 __bf16* __restrict__ Aswz, int n) {
    const int idx = blockIdx.x * blockDim.x + threadIdx.x;
    if (idx >= n * HDIM) return;
    Aswz[aswz_pos(idx >> 7, idx & 127)] = (__bf16)Hm[idx];
}

// W[128,128] fp32 -> Wswz fragment order: ((colTile*4 + s)*32 + lane)*16 + e
// B-matrix 32x16 bf16: lane = n | hi<<4, k = s*32 + (e>>3)*16 + hi*8 + (e&7).
__global__ void swizzle_W_kernel(const float* __restrict__ W,
                                 __bf16* __restrict__ Wswz) {
    const int idx = blockIdx.x * blockDim.x + threadIdx.x;   // 8*4*32*16 = 16384
    if (idx >= 8 * 4 * 32 * 16) return;
    const int e    = idx & 15;
    const int lane = (idx >> 4) & 31;
    const int s    = (idx >> 9) & 3;
    const int c    = idx >> 11;
    const int n    = lane & 15, hi = lane >> 4;
    const int k    = s * 32 + (e >> 3) * 16 + hi * 8 + (e & 7);
    Wswz[idx] = (__bf16)W[k * HDIM + c * 16 + n];
}

// ----------------------- node feature transform (WMMA) ---------------------
// 256-thread block (8 waves) computes 64 rows x 128 cols.  Wave w owns output
// columns [16w,16w+16); 4 row-tiles per wave share each B fragment.
// All operands are pre-swizzled bf16 -> pure global_load_b128 fragment loads.
__global__ void __launch_bounds__(256)
gemm_nodes_wmma(const __bf16* __restrict__ Aswz, const __bf16* __restrict__ Wswz,
                float* __restrict__ Out, int Nrows) {
    const int tid  = threadIdx.x;
    const int wave = tid >> 5;
    const int lane = tid & 31;
    const int tile0 = blockIdx.x * 4;       // first 16-row tile of this block
    const int mn = lane & 15;
    const int hi = (lane >> 4) & 1;

    v8f acc[4] = {{}, {}, {}, {}};
    #pragma unroll
    for (int s = 0; s < 4; ++s) {           // K-steps of 32
        const v16bf b = *(const v16bf*)(Wswz + (((size_t)wave * 4 + s) * 32 + lane) * 16);
        #pragma unroll
        for (int t = 0; t < 4; ++t) {
            const v16bf a = *(const v16bf*)(
                Aswz + ((((size_t)(tile0 + t)) * 4 + s) * 32 + lane) * 16);
            acc[t] = __builtin_amdgcn_wmma_f32_16x16x32_bf16(
                         false, a, false, b, (short)0, acc[t], false, false);
        }
    }

    // D 16x16 f32 layout: VGPR r -> row r + 8*hi, lane -> column
    const int n0 = wave << 4;
    #pragma unroll
    for (int t = 0; t < 4; ++t) {
        #pragma unroll
        for (int r = 0; r < 8; ++r) {
            const int gr = (tile0 + t) * 16 + r + (hi << 3);
            if (gr < Nrows) Out[(size_t)gr * HDIM + n0 + mn] = acc[t][r];
        }
    }
}

// ------------------------------ edge scatter --------------------------------
// 32 lanes per edge; float4 gather of source row, 4 global_atomic_add_f32
// into the destination row (both buffers L2-resident on MI455X's 192MB L2).
__global__ void scatter_edges(const float* __restrict__ msg,
                              const int* __restrict__ src,
                              const int* __restrict__ dst,
                              const float* __restrict__ dinv,
                              float* __restrict__ agg, int E) {
    const int t = blockIdx.x * blockDim.x + threadIdx.x;
    const int e = t >> 5;
    const int lane = t & 31;
    if (e >= E) return;
    const int s = src[e], d = dst[e];
    if (e + 1 < E) {   // warm L2/L0 for the next edge's source row
        __builtin_prefetch(msg + (size_t)src[e + 1] * HDIM + (lane << 2), 0, 1);
    }
    const float w = dinv[s] * dinv[d];
    const float4 v = ((const float4*)(msg + (size_t)s * HDIM))[lane];
    float* row = agg + (size_t)d * HDIM + (lane << 2);
    atomicAdd(row + 0, v.x * w);
    atomicAdd(row + 1, v.y * w);
    atomicAdd(row + 2, v.z * w);
    atomicAdd(row + 3, v.w * w);
}

// ---------- fused self-loop + bias + BatchNorm + ReLU (+ bf16 swizzle) ------
__global__ void epilogue_bn_relu(const float* __restrict__ msg,
                                 float* __restrict__ agg,
                                 __bf16* __restrict__ Aswz,
                                 const float* __restrict__ dinv,
                                 const float* __restrict__ bias,
                                 const float* __restrict__ gamma,
                                 const float* __restrict__ beta,
                                 const float* __restrict__ mean,
                                 const float* __restrict__ var,
                                 int n) {
    const int idx = blockIdx.x * blockDim.x + threadIdx.x;
    if (idx >= n * HDIM) return;
    const int i = idx >> 7, f = idx & 127;
    const float di = dinv[i];
    float v = agg[idx] + msg[idx] * di * di + bias[f];          // agg + self-loop + b
    v = (v - mean[f]) * rsqrtf(var[f] + 1e-5f) * gamma[f] + beta[f];
    v = fmaxf(v, 0.0f);
    agg[idx] = v;                        // fp32 h (scatter acc reuse / pooling)
    Aswz[aswz_pos(i, f)] = (__bf16)v;    // next layer's WMMA A operand
}

// ----------------------------- mean pooling ---------------------------------
__global__ void pool_scatter(const float* __restrict__ h,
                             const int* __restrict__ batch,
                             float* __restrict__ gsum,
                             float* __restrict__ cnt, int n) {
    const int t = blockIdx.x * blockDim.x + threadIdx.x;
    const int i = t >> 5, lane = t & 31;
    if (i >= n) return;
    const int g = batch[i];
    const float4 v = ((const float4*)(h + (size_t)i * HDIM))[lane];
    float* row = gsum + (size_t)g * HDIM + (lane << 2);
    atomicAdd(row + 0, v.x);
    atomicAdd(row + 1, v.y);
    atomicAdd(row + 2, v.z);
    atomicAdd(row + 3, v.w);
    if (lane == 0) atomicAdd(&cnt[g], 1.0f);
}

// z[G,192] = [ gsum/cnt  |  relu(scalar @ Ws + bs) ]
__global__ void build_z(const float* __restrict__ gsum, const float* __restrict__ cnt,
                        const float* __restrict__ scalar, const float* __restrict__ Ws,
                        const float* __restrict__ bs, float* __restrict__ z, int G) {
    const int idx = blockIdx.x * blockDim.x + threadIdx.x;
    if (idx >= G * (HDIM + SHALF)) return;
    const int g = idx / (HDIM + SHALF);
    const int j = idx % (HDIM + SHALF);
    if (j < HDIM) {
        z[idx] = gsum[(size_t)g * HDIM + j] / fmaxf(cnt[g], 1.0f);
    } else {
        const int j2 = j - HDIM;
        float acc = bs[j2];
        #pragma unroll
        for (int k = 0; k < 8; ++k) acc += scalar[g * 8 + k] * Ws[k * SHALF + j2];
        z[idx] = fmaxf(acc, 0.0f);
    }
}

// ------------------------------ classifier ----------------------------------
__global__ void fc1_relu(const float* __restrict__ z, const float* __restrict__ Wc1,
                         const float* __restrict__ bc1, float* __restrict__ z2, int G) {
    const int idx = blockIdx.x * blockDim.x + threadIdx.x;
    if (idx >= G * HDIM) return;
    const int g = idx >> 7, j = idx & 127;
    float acc = bc1[j];
    for (int k = 0; k < HDIM + SHALF; ++k)
        acc += z[g * (HDIM + SHALF) + k] * Wc1[k * HDIM + j];
    z2[idx] = fmaxf(acc, 0.0f);
}

__global__ void fc2(const float* __restrict__ z2, const float* __restrict__ Wc2,
                    const float* __restrict__ bc2, float* __restrict__ out, int G) {
    const int idx = blockIdx.x * blockDim.x + threadIdx.x;
    if (idx >= G * CDIM) return;
    const int g = idx >> 4, c = idx & 15;
    float acc = bc2[c];
    #pragma unroll 8
    for (int k = 0; k < HDIM; ++k) acc += z2[g * HDIM + k] * Wc2[k * CDIM + c];
    out[idx] = acc;
}

// ---------------------------------------------------------------------------
extern "C" void kernel_launch(void* const* d_in, const int* in_sizes, int n_in,
                              void* d_out, int out_size, void* d_ws, size_t ws_size,
                              hipStream_t stream) {
    const float* x      = (const float*)d_in[0];
    const int*   edge   = (const int*)  d_in[1];
    const int*   batch  = (const int*)  d_in[2];
    const float* scalar = (const float*)d_in[3];
    const float* W0     = (const float*)d_in[4];
    const float* b0     = (const float*)d_in[5];
    const float* Wk     = (const float*)d_in[6];
    const float* bk     = (const float*)d_in[7];
    const float* gamma  = (const float*)d_in[8];
    const float* beta   = (const float*)d_in[9];
    const float* mean   = (const float*)d_in[10];
    const float* var    = (const float*)d_in[11];
    const float* Ws     = (const float*)d_in[12];
    const float* bs     = (const float*)d_in[13];
    const float* Wc1    = (const float*)d_in[14];
    const float* bc1    = (const float*)d_in[15];
    const float* Wc2    = (const float*)d_in[16];
    const float* bc2    = (const float*)d_in[17];

    const int N = in_sizes[0] / HDIM;   // 50000
    const int E = in_sizes[1] / 2;      // 800000
    const int G = in_sizes[3] / 8;      // 256
    const int* src = edge;
    const int* dst = edge + E;

    const int nTiles  = (N + 15) / 16;          // 3125
    const int gBlocks = (nTiles + 3) / 4;       // 782 (64 rows / block)

    // ---------------- workspace layout (256B-aligned byte regions) ----------
    char* base = (char*)d_ws;
    size_t off = 0;
    auto alloc = [&](size_t bytes) {
        off = (off + 255) & ~(size_t)255;
        void* p = base + off;
        off += bytes;
        return p;
    };
    float*  dinv = (float*) alloc((size_t)N * 4);              // deg -> rsqrt(deg)
    float*  tmp  = (float*) alloc((size_t)N * HDIM * 4);       // h @ W
    float*  bufA = (float*) alloc((size_t)N * HDIM * 4);       // scatter acc / h
    __bf16* Aswz = (__bf16*)alloc((size_t)gBlocks * 4 * 2048 * 2); // padded tiles
    __bf16* Wswz = (__bf16*)alloc((size_t)3 * 16384 * 2);
    float*  gsum = (float*) alloc((size_t)G * HDIM * 4);
    float*  cnt  = (float*) alloc((size_t)G * 4);
    float*  z    = (float*) alloc((size_t)G * (HDIM + SHALF) * 4);
    float*  z2   = (float*) alloc((size_t)G * HDIM * 4);
    (void)ws_size; (void)n_in; (void)out_size;

    const int T = 256;
    auto blocks = [](long long n, int t) { return (int)((n + t - 1) / t); };

    // ---------------- degree + symmetric norm -------------------------------
    set_ones_f32<<<blocks(N, T), T, 0, stream>>>(dinv, N);        // self-loops
    deg_accum<<<blocks(E, T), T, 0, stream>>>(dst, dinv, E);
    finalize_dinv<<<blocks(N, T), T, 0, stream>>>(dinv, N);

    // ---------------- one-time operand swizzles -----------------------------
    swizzle_x_kernel<<<blocks((long long)N * HDIM, T), T, 0, stream>>>(x, Aswz, N);
    swizzle_W_kernel<<<blocks(16384, T), T, 0, stream>>>(W0, Wswz);
    swizzle_W_kernel<<<blocks(16384, T), T, 0, stream>>>(Wk, Wswz + 16384);
    swizzle_W_kernel<<<blocks(16384, T), T, 0, stream>>>(Wk + (size_t)HDIM * HDIM,
                                                         Wswz + 2 * 16384);

    // ---------------- 3x (GCN transform + aggregate + BN + ReLU) ------------
    const float* layer_b[3] = { b0, bk, bk + HDIM };
    for (int l = 0; l < 3; ++l) {
        gemm_nodes_wmma<<<gBlocks, 256, 0, stream>>>(Aswz, Wswz + (size_t)l * 16384,
                                                     tmp, N);
        zero_f32<<<blocks((long long)N * HDIM, T), T, 0, stream>>>(bufA, N * HDIM);
        scatter_edges<<<blocks((long long)E * 32, T), T, 0, stream>>>(
            tmp, src, dst, dinv, bufA, E);
        epilogue_bn_relu<<<blocks((long long)N * HDIM, T), T, 0, stream>>>(
            tmp, bufA, Aswz, dinv, layer_b[l],
            gamma + l * HDIM, beta + l * HDIM, mean + l * HDIM, var + l * HDIM, N);
    }

    // ---------------- pooling + scalar branch + classifier ------------------
    zero_f32<<<blocks((long long)G * HDIM, T), T, 0, stream>>>(gsum, G * HDIM);
    zero_f32<<<blocks(G, T), T, 0, stream>>>(cnt, G);
    pool_scatter<<<blocks((long long)N * 32, T), T, 0, stream>>>(bufA, batch, gsum, cnt, N);
    build_z<<<blocks((long long)G * (HDIM + SHALF), T), T, 0, stream>>>(
        gsum, cnt, scalar, Ws, bs, z, G);
    fc1_relu<<<blocks((long long)G * HDIM, T), T, 0, stream>>>(z, Wc1, bc1, z2, G);
    fc2<<<blocks((long long)G * CDIM, T), T, 0, stream>>>(z2, Wc2, bc2, (float*)d_out, G);
}
